// BIG_Fastfood_Layer_72799695667730
// MI455X (gfx1250) — matrix operand order repaired and checked by reading, not compiled, not run on hardware
//
#include <hip/hip_runtime.h>
#include <hip/hip_bf16.h>

typedef _Float16 v16h __attribute__((ext_vector_type(16)));
typedef _Float16 v8h  __attribute__((ext_vector_type(8)));
typedef float    v8f  __attribute__((ext_vector_type(8)));

#define NW       8        // waves per workgroup (one x-row per wave)
#define D_IN     1024
#define LSTRIDE  40       // f16 staging row stride: 80B -> 16B aligned, bank-friendly

__device__ __forceinline__ float hadsign(int i, int j) {
    return (__popc(i & j) & 1) ? -1.0f : 1.0f;
}

// One full 1024-point FWHT:  Z = H32 * T * H32, T given as B-operand fragments.
// bf[jh][e] = T[kb+e][jh*16+n]  (kb = (lane>>4)*16, n = lane&15)
// Result left in myF (flat, f32, myF[r*32+c] = Z[r][c]).
__device__ __forceinline__ void fwht_wmma(const v16h bf[2], const v16h hA[2], const v16h hB[2],
                                          _Float16* __restrict__ myH, float* __restrict__ myF,
                                          int n, int khalf) {
    // ---- stage 1: Y = H * T  (A = const Hadamard rows, B = data) ----
    #pragma unroll
    for (int ih = 0; ih < 2; ++ih) {
        #pragma unroll
        for (int jh = 0; jh < 2; ++jh) {
            v8f acc = {};
            acc = __builtin_amdgcn_wmma_f32_16x16x32_f16(
                false, hA[ih], false, bf[jh], (short)0, acc, false, false);
            // D tile (f32) -> f16 staging buffer, row-major with padded stride
            #pragma unroll
            for (int v = 0; v < 8; ++v) {
                int r = ih * 16 + khalf * 8 + v;
                int c = jh * 16 + n;
                myH[r * LSTRIDE + c] = (_Float16)acc[v];
            }
        }
    }
    __builtin_amdgcn_wave_barrier();   // same-wave LDS is in-order; pin compiler order

    // ---- stage 2: Z = Y * H  (A = Y from LDS, B = const Hadamard cols) ----
    #pragma unroll
    for (int ih = 0; ih < 2; ++ih) {
        int r = ih * 16 + n;           // A-layout: lane's row = lane&15 within tile
        v8h a0 = *(const v8h*)&myH[r * LSTRIDE + khalf * 16];       // K = khalf*16 + 0..7
        v8h a1 = *(const v8h*)&myH[r * LSTRIDE + khalf * 16 + 8];   // K = khalf*16 + 8..15
        v16h af;
        #pragma unroll
        for (int e = 0; e < 8; ++e) { af[e] = a0[e]; af[8 + e] = a1[e]; }
        #pragma unroll
        for (int jh = 0; jh < 2; ++jh) {
            v8f acc = {};
            acc = __builtin_amdgcn_wmma_f32_16x16x32_f16(
                false, af, false, hB[jh], (short)0, acc, false, false);
            #pragma unroll
            for (int v = 0; v < 8; ++v) {
                int r2 = ih * 16 + khalf * 8 + v;
                int c2 = jh * 16 + n;
                myF[r2 * 32 + c2] = acc[v];
            }
        }
    }
    __builtin_amdgcn_wave_barrier();
}

__global__ __launch_bounds__(NW * 32)
void fastfood_fused_kernel(const float* __restrict__ x, const float* __restrict__ Bp,
                           const float* __restrict__ Gp, const float* __restrict__ Sp,
                           const int*   __restrict__ Pp, const float* __restrict__ Up,
                           float* __restrict__ out) {
    __shared__ __align__(16) _Float16 ldsH[NW][32 * LSTRIDE]; // Y staging (f16)
    __shared__ __align__(16) float    ldsF[NW][D_IN];         // FWHT result / epilogue staging

    const int lane  = threadIdx.x & 31;
    const int w     = threadIdx.x >> 5;
    const int row   = blockIdx.x * NW + w;
    const int n     = lane & 15;
    const int khalf = lane >> 4;
    const int kb    = khalf * 16;

    _Float16* myH = ldsH[w];
    float*    myF = ldsF[w];

    // ---- constant Hadamard fragments (exact in f16: entries are +/-1) ----
    v16h hA[2], hB[2];
    #pragma unroll
    for (int ih = 0; ih < 2; ++ih)
        #pragma unroll
        for (int e = 0; e < 16; ++e) {
            int k = kb + e;
            hA[ih][e] = (_Float16)hadsign(ih * 16 + n, k);
        }
    #pragma unroll
    for (int jh = 0; jh < 2; ++jh)
        #pragma unroll
        for (int e = 0; e < 16; ++e)
            hB[jh][e] = (_Float16)hadsign(kb + e, jh * 16 + n);

    // ---- cache this row of x in registers (reused across all 8 blocks) ----
    // lane owns flat indices j = (kb+e)*32 + jh*16 + n  == its B-fragment slots
    float xr[2][16];
    const float* xrow = x + (size_t)row * D_IN;
    #pragma unroll
    for (int jh = 0; jh < 2; ++jh)
        #pragma unroll
        for (int e = 0; e < 16; ++e)
            xr[jh][e] = xrow[(kb + e) * 32 + jh * 16 + n];

    const float inv_sqrt_d = 0.03125f;               // 1/sqrt(1024)
    const float inv2pi     = 0.15915494309189535f;   // v_cos_f32 takes turns
    const float amp        = 0.015625f;              // sqrt(2/8192) = 1/64

    for (int m = 0; m < 8; ++m) {
        const float* Bm = Bp + m * D_IN;
        const float* Gm = Gp + m * D_IN;
        const float* Sm = Sp + m * D_IN;
        const int*   Pm = Pp + m * D_IN;
        const float* Um = Up + m * D_IN;

        if (m < 7) {   // global_prefetch_b8 next block's params
            __builtin_prefetch(Bm + D_IN, 0, 0);
            __builtin_prefetch(Gm + D_IN, 0, 0);
            __builtin_prefetch(Sm + D_IN, 0, 0);
            __builtin_prefetch((const void*)(Pm + D_IN), 0, 0);
            __builtin_prefetch(Um + D_IN, 0, 0);
        }

        // ---- t = B o x, packed straight into B-operand fragments ----
        v16h bf[2];
        #pragma unroll
        for (int jh = 0; jh < 2; ++jh)
            #pragma unroll
            for (int e = 0; e < 16; ++e) {
                int j = (kb + e) * 32 + jh * 16 + n;
                bf[jh][e] = (_Float16)(xr[jh][e] * Bm[j]);
            }

        fwht_wmma(bf, hA, hB, myH, myF, n, khalf);   // FWHT #1 -> myF

        // ---- permutation gather + G, packed into next B fragments ----
        v16h bf2[2];
        #pragma unroll
        for (int jh = 0; jh < 2; ++jh)
            #pragma unroll
            for (int e = 0; e < 16; ++e) {
                int j = (kb + e) * 32 + jh * 16 + n;
                bf2[jh][e] = (_Float16)(myF[Pm[j]] * Gm[j]);
            }
        __builtin_amdgcn_wave_barrier();

        fwht_wmma(bf2, hA, hB, myH, myF, n, khalf);  // FWHT #2 -> myF

        // ---- epilogue: S-scale, cos(v/32 + 2*pi*u)/64, fully coalesced ----
        float* orow = out + (size_t)row * 8192 + m * D_IN;
        #pragma unroll 4
        for (int e = 0; e < 32; ++e) {
            int j = e * 32 + lane;
            float v = myF[j] * Sm[j] * inv_sqrt_d;
            float t = __builtin_fmaf(v, inv2pi, Um[j]);   // turns
            orow[j] = __builtin_amdgcn_cosf(t) * amp;
        }
        __builtin_amdgcn_wave_barrier();
    }
}

extern "C" void kernel_launch(void* const* d_in, const int* in_sizes, int n_in,
                              void* d_out, int out_size, void* d_ws, size_t ws_size,
                              hipStream_t stream) {
    const float* x  = (const float*)d_in[0];
    const float* B  = (const float*)d_in[1];
    const float* G  = (const float*)d_in[2];
    const float* S  = (const float*)d_in[3];
    const int*   P  = (const int*)d_in[4];
    const float* u  = (const float*)d_in[5];
    float* out = (float*)d_out;

    dim3 grid(8192 / NW);      // one row per wave, NW waves per workgroup
    dim3 block(NW * 32);
    fastfood_fused_kernel<<<grid, block, 0, stream>>>(x, B, G, S, P, u, out);
}
